// FASTMultiHeadAttention_31078383354363
// MI455X (gfx1250) — matrix operand order repaired and checked by reading, not compiled, not run on hardware
//
#include <hip/hip_runtime.h>

// fastmax (degree-2 Taylor softmax attention) with RPE, flash-style, f16 WMMA.
// B=2,H=8 -> BH=16; N=1024; D=64. Output fp32 (B,H,N,D).

#define BH_TOT 16
#define SEQ    1024
#define DH     64
#define RPE_LAST (2 * SEQ - 2)     // last valid rpe row index

#define WAVES           4
#define ROWS_PER_BLOCK  64          // WAVES * 16
#define PSTRIDE         1040        // per-row P entries (t = a + N-1 - j in [0,1038]), padded
#define PTILES          65          // 65*16 = 1040
#define JT              32          // columns (j) per main-loop step

typedef __attribute__((ext_vector_type(16))) _Float16 v16h;
typedef __attribute__((ext_vector_type(8)))  _Float16 v8h;
typedef __attribute__((ext_vector_type(8)))  float    v8f;
typedef __attribute__((ext_vector_type(4)))  float    v4f;

#define WMMA_F32_F16(A, B, C) \
  __builtin_amdgcn_wmma_f32_16x16x32_f16(false, (A), false, (B), (short)0, (C), false, false)

// Convert 16 contiguous f32 -> v16h (B-tile fragment: element e <-> K = e + 16*g).
__device__ __forceinline__ v16h cvt16_from_f32(const float* __restrict__ src) {
  const v4f* p = (const v4f*)src;
  v4f f0 = p[0], f1 = p[1], f2 = p[2], f3 = p[3];
  v16h r;
#pragma unroll
  for (int i = 0; i < 4; ++i) {
    r[i]      = (_Float16)f0[i];
    r[4 + i]  = (_Float16)f1[i];
    r[8 + i]  = (_Float16)f2[i];
    r[12 + i] = (_Float16)f3[i];
  }
  return r;
}

// Load 16 contiguous f16 from LDS (16B-aligned) -> v16h.
__device__ __forceinline__ v16h lds16(const _Float16* p) {
  v8h lo = *(const v8h*)p;
  v8h hi = *(const v8h*)(p + 8);
  v16h r;
#pragma unroll
  for (int i = 0; i < 8; ++i) { r[i] = lo[i]; r[8 + i] = hi[i]; }
  return r;
}

__global__ __launch_bounds__(128) void fastmax_wmma_kernel(
    const float* __restrict__ q, const float* __restrict__ k,
    const float* __restrict__ v, const float* __restrict__ rpe,
    float* __restrict__ out) {
  // Static LDS: 4*16*1040 + 32*64 + 64*32 + 4*16*32 halves = 142 KB (<= 320 KB/WG)
  __shared__ __align__(16) _Float16 Plds[WAVES * 16 * PSTRIDE];  // P[a][t] per wave
  __shared__ __align__(16) _Float16 Klds[JT * DH];               // K tile, row-major [j][d]
  __shared__ __align__(16) _Float16 Vtld[DH * JT];               // V tile transposed [d][j]
  __shared__ __align__(16) _Float16 Slds[WAVES * 16 * 32];       // S tile per wave [a][b]

  const int tid  = threadIdx.x;
  const int wid  = tid >> 5;
  const int lane = tid & 31;
  const int g    = lane >> 4;   // lane group (0: lanes 0-15, 1: lanes 16-31)
  const int ln   = lane & 15;   // row m (A role) or col n (B/C role)

  const int bh = blockIdx.y;
  const int i0 = blockIdx.x * ROWS_PER_BLOCK + wid * 16;

  const float* qb = q + (size_t)bh * SEQ * DH;
  const float* kb = k + (size_t)bh * SEQ * DH;
  const float* vb = v + (size_t)bh * SEQ * DH;
  float*       ob = out + (size_t)bh * SEQ * DH;

  // ---- Q tile in A layout (row m = ln): element e <-> k = 16g + e ----
  v16h aq0, aq1;
  {
    const float* qrow = qb + (size_t)(i0 + ln) * DH;
    const v4f* p;
    v4f f0, f1, f2, f3;
    p = (const v4f*)(qrow + 16 * g);      f0 = p[0]; f1 = p[1];  // k = 16g..16g+7
    p = (const v4f*)(qrow + 8 + 16 * g);  f2 = p[0]; f3 = p[1];  // k = 8+16g..+7
#pragma unroll
    for (int i = 0; i < 4; ++i) {
      aq0[i]      = (_Float16)f0[i];
      aq0[4 + i]  = (_Float16)f1[i];
      aq0[8 + i]  = (_Float16)f2[i];
      aq0[12 + i] = (_Float16)f3[i];
    }
    p = (const v4f*)(qrow + 32 + 16 * g); f0 = p[0]; f1 = p[1];  // k = 32+16g..
    p = (const v4f*)(qrow + 40 + 16 * g); f2 = p[0]; f3 = p[1];  // k = 40+16g..
#pragma unroll
    for (int i = 0; i < 4; ++i) {
      aq1[i]      = (_Float16)f0[i];
      aq1[4 + i]  = (_Float16)f1[i];
      aq1[8 + i]  = (_Float16)f2[i];
      aq1[12 + i] = (_Float16)f3[i];
    }
  }

  // ---- Prologue: P[a][t] = q_{i0+a} . rpe[i0+t], t in [0,1040) ----
  _Float16* Pw = Plds + wid * (16 * PSTRIDE);
#pragma unroll 1
  for (int nt = 0; nt < PTILES; ++nt) {
    int r = i0 + nt * 16 + ln;            // B-tile column t = nt*16 + ln -> rpe row i0+t
    if (r > RPE_LAST) r = RPE_LAST;       // only padded t=1039 can overflow; value unused
    const float* rrow = rpe + (size_t)r * DH;
    v16h b0 = cvt16_from_f32(rrow + 16 * g);        // d = 16g + e       (K chunk 0)
    v16h b1 = cvt16_from_f32(rrow + 32 + 16 * g);   // d = 32 + 16g + e  (K chunk 1)
    v8f c = {};
    c = WMMA_F32_F16(aq0, b0, c);
    c = WMMA_F32_F16(aq1, b1, c);
#pragma unroll
    for (int vv = 0; vv < 8; ++vv)
      Pw[(vv + 8 * g) * PSTRIDE + nt * 16 + ln] = (_Float16)c[vv];
  }
  __syncthreads();

  v8f oacc0 = {}, oacc1 = {}, oacc2 = {}, oacc3 = {};
  v8f dacc = {};
  _Float16* Sw = Slds + wid * (16 * 32);

#pragma unroll 1
  for (int j0 = 0; j0 < SEQ; j0 += JT) {
    // ---- cooperative staging: K tile (row-major f16) and V tile (transposed f16) ----
    {
      const int row = tid >> 2;             // 0..31
      const int c0  = (tid & 3) * 16;       // 0,16,32,48
      const v4f* kp = (const v4f*)(kb + (size_t)(j0 + row) * DH + c0);
      const v4f* vp = (const v4f*)(vb + (size_t)(j0 + row) * DH + c0);
      v4f kf[4], vf[4];
#pragma unroll
      for (int i = 0; i < 4; ++i) { kf[i] = kp[i]; vf[i] = vp[i]; }
#pragma unroll
      for (int i = 0; i < 16; ++i) {
        Klds[row * DH + c0 + i]   = (_Float16)kf[i >> 2][i & 3];
        Vtld[(c0 + i) * JT + row] = (_Float16)vf[i >> 2][i & 3];
      }
    }
    __syncthreads();

    // ---- S = Q K^T + RPE-gather; s = 1 + t + 0.5 t^2 ; stage S (f16, A-orientation via LDS) ----
#pragma unroll
    for (int sub = 0; sub < 2; ++sub) {
      const _Float16* krow = Klds + (sub * 16 + ln) * DH;   // B col n = ln -> K row j0+16sub+ln
      v16h bk0 = lds16(krow + 16 * g);                      // k = 16g + e
      v16h bk1 = lds16(krow + 32 + 16 * g);                 // k = 32 + 16g + e
      v8f c = {};
      c = WMMA_F32_F16(aq0, bk0, c);
      c = WMMA_F32_F16(aq1, bk1, c);
      const int j = j0 + sub * 16 + ln;
#pragma unroll
      for (int vv = 0; vv < 8; ++vv) {
        const int a = vv + 8 * g;                           // C row
        float t = c[vv] + (float)Pw[a * PSTRIDE + (a + (SEQ - 1) - j)];
        float s = 1.0f + t + 0.5f * t * t;                  // Taylor kernel score
        dacc[vv] += s;                                      // row-sum (denominator)
        Sw[a * 32 + sub * 16 + ln] = (_Float16)s;
      }
    }
    asm volatile("" ::: "memory");  // keep S stores before A-layout reloads (wave-private LDS)

    // ---- reload S as A operand: element e <-> b = 16g + e, row m = ln ----
    v16h as;
    {
      v8h lo = *(const v8h*)(Sw + ln * 32 + 16 * g);
      v8h hi = *(const v8h*)(Sw + ln * 32 + 8 + 16 * g);
#pragma unroll
      for (int i = 0; i < 8; ++i) { as[i] = lo[i]; as[8 + i] = hi[i]; }
    }

    // ---- O += S * V : B[b][d] = Vt[d][b], contiguous per lane ----
    {
      v16h bv;
      bv = lds16(Vtld + (0 * 16 + ln) * JT + 16 * g); oacc0 = WMMA_F32_F16(as, bv, oacc0);
      bv = lds16(Vtld + (1 * 16 + ln) * JT + 16 * g); oacc1 = WMMA_F32_F16(as, bv, oacc1);
      bv = lds16(Vtld + (2 * 16 + ln) * JT + 16 * g); oacc2 = WMMA_F32_F16(as, bv, oacc2);
      bv = lds16(Vtld + (3 * 16 + ln) * JT + 16 * g); oacc3 = WMMA_F32_F16(as, bv, oacc3);
    }
    __syncthreads();  // protect K/V/S LDS before next iteration's staging
  }

  // ---- denominator: reduce row-sums across the 16 lanes of each half-wave ----
#pragma unroll
  for (int off = 1; off < 16; off <<= 1) {
#pragma unroll
    for (int vv = 0; vv < 8; ++vv)
      dacc[vv] += __shfl_xor(dacc[vv], off, 32);
  }

  // ---- write O / denom ; C layout: lane (g, ln) holds rows vv+8g, col d = 16*nt + ln ----
#pragma unroll
  for (int vv = 0; vv < 8; ++vv) {
    const float rinv = 1.0f / dacc[vv];
    float* orow = ob + (size_t)(i0 + vv + 8 * g) * DH;
    orow[0 * 16 + ln] = oacc0[vv] * rinv;
    orow[1 * 16 + ln] = oacc1[vv] * rinv;
    orow[2 * 16 + ln] = oacc2[vv] * rinv;
    orow[3 * 16 + ln] = oacc3[vv] * rinv;
  }
}

extern "C" void kernel_launch(void* const* d_in, const int* in_sizes, int n_in,
                              void* d_out, int out_size, void* d_ws, size_t ws_size,
                              hipStream_t stream) {
  (void)in_sizes; (void)n_in; (void)out_size; (void)d_ws; (void)ws_size;
  const float* q   = (const float*)d_in[0];
  const float* k   = (const float*)d_in[1];
  const float* v   = (const float*)d_in[2];
  // d_in[3] = drop_noise (unused: dropout = 0.0 in reference)
  const float* rpe = (const float*)d_in[4];
  float* out = (float*)d_out;

  dim3 grid(SEQ / ROWS_PER_BLOCK, BH_TOT, 1);   // 16 x 16 = 256 blocks
  dim3 block(128, 1, 1);                        // 4 wave32 waves
  fastmax_wmma_kernel<<<grid, block, 0, stream>>>(q, k, v, rpe, out);
}
